// TCWinBlock_81990925681418
// MI455X (gfx1250) — matrix operand order repaired and checked by reading, not compiled, not run on hardware
//
#include <hip/hip_runtime.h>

// ---------------------------------------------------------------------------
// TCFormer-style block for MI455X (gfx1250, wave32, WMMA + TDM).
// GEMMs: bf16 operands, fp32 accumulate via v_wmma_f32_16x16x32_bf16,
// tiles moved by the Tensor Data Mover (tensor_load_to_lds, TENSORcnt).
// ---------------------------------------------------------------------------

#define BB    8
#define HH    56
#define WW2   56
#define NN    (HH * WW2)          // 3136
#define CC    256
#define HID   1024
#define HEADS 8
#define HD    32
#define WS    7
#define WS2   49
#define NTOK  (BB * NN)           // 25088

typedef __attribute__((ext_vector_type(16))) __bf16       v16bf;
typedef __attribute__((ext_vector_type(8)))  float        v8f;
typedef __attribute__((ext_vector_type(4)))  unsigned int u32x4;
typedef __attribute__((ext_vector_type(8)))  int          i32x8;
typedef __attribute__((ext_vector_type(4)))  int          i32x4;

#ifndef __has_builtin
#define __has_builtin(x) 0
#endif
#if __has_builtin(__builtin_amdgcn_tensor_load_to_lds) && \
    __has_builtin(__builtin_amdgcn_s_wait_tensorcnt)
#define USE_TDM 1
#else
#define USE_TDM 0
#endif

union FragBF {
    v16bf v;
    u32x4 q[2];
    unsigned short s[16];
};

__device__ __forceinline__ unsigned short f2bf(float f) {
    unsigned int u = __float_as_uint(f);
    u += 0x7FFFu + ((u >> 16) & 1u);          // round-to-nearest-even
    return (unsigned short)(u >> 16);
}
__device__ __forceinline__ float bf2f(unsigned short s) {
    return __uint_as_float(((unsigned int)s) << 16);
}
__device__ __forceinline__ float gelu_f(float x) {
    float x3 = x * x * x;
    return 0.5f * x * (1.f + tanhf(0.7978845608028654f * (x + 0.044715f * x3)));
}

// --- WMMA fragment loads from LDS tiles (bf16, K=32 step) ------------------
// A (16x32, MxK): lane&15 = m; halves 0..7 -> K = (lane<16?0:8)+h,
//                 halves 8..15 -> K = (lane<16?16:24)+(h-8)   (ISA 7.12.2)
__device__ __forceinline__ FragBF load_a_frag(const unsigned short* tile,
                                              int stride, int rowBase,
                                              int kBase, int lane) {
    FragBF f;
    int m   = rowBase + (lane & 15);
    int klo = kBase + ((lane & 16) ? 8 : 0);
    f.q[0] = *(const u32x4*)(tile + m * stride + klo);
    f.q[1] = *(const u32x4*)(tile + m * stride + klo + 16);
    return f;
}
// B (32x16, KxN) staged as tile[n][k]: lane&15 = n; halves 0..7 -> K =
// (lane<16?0:8)+h, halves 8..15 -> K = (lane<16?16:24)+(h-8) (matches A).
__device__ __forceinline__ FragBF load_b_frag(const unsigned short* tile,
                                              int stride, int nBase,
                                              int kBase, int lane) {
    FragBF f;
    int n  = nBase + (lane & 15);
    int k0 = kBase + ((lane & 16) ? 8 : 0);
    f.q[0] = *(const u32x4*)(tile + n * stride + k0);
    f.q[1] = *(const u32x4*)(tile + n * stride + k0 + 16);
    return f;
}

#define WMMA_BF16(A_, B_, C_)                                                  \
    __builtin_amdgcn_wmma_f32_16x16x32_bf16(false, (A_), false, (B_),          \
                                            (short)0, (C_), false, false)

// --- Tensor Data Mover: 2D bf16 tile (32 elem wide x tile_rows) -> LDS -----
// D# bitfields per CDNA5 ISA ch.8: group0 = {count, lds_addr, global_addr,
// type=2}; group1 = {data_size=2B, tensor_dim0=row_len, tensor_dim1=rows,
// tile_dim0=32, tile_dim1=tile_rows, tensor_dim0_stride=row_len}.
// 6-arg builtin form (clang-23/therock): (g0, g1, g2, g3, g_extra, cpol).
__device__ __forceinline__ void tdm_load_tile_2d(const unsigned short* g,
                                                 unsigned lds_off,
                                                 unsigned tensor_rows,
                                                 unsigned row_len,
                                                 unsigned tile_rows) {
#if USE_TDM
    unsigned long long ga = (unsigned long long)(size_t)g;
    u32x4 g0;
    g0[0] = 1u;                                                  // count=1
    g0[1] = lds_off;                                             // lds_addr
    g0[2] = (unsigned)ga;                                        // gaddr[31:0]
    g0[3] = (unsigned)((ga >> 32) & 0x01ffffffu) | 0x80000000u;  // gaddr[56:32]|type=2
    i32x8 g1;
    g1[0] = 1 << 16;                                             // data_size=2B
    g1[1] = (int)((row_len & 0xffffu) << 16);                    // t_dim0[15:0]
    g1[2] = (int)((row_len >> 16) | ((tensor_rows & 0xffffu) << 16));
    g1[3] = (int)((tensor_rows >> 16) | (32u << 16));            // tile_dim0=32
    g1[4] = (int)(tile_rows & 0xffffu);                          // tile_dim1
    g1[5] = (int)row_len;                                        // d0_stride
    g1[6] = 0;
    g1[7] = 0;
    i32x4 z4 = {0, 0, 0, 0};
    i32x8 z8 = {0, 0, 0, 0, 0, 0, 0, 0};
    __builtin_amdgcn_tensor_load_to_lds(g0, g1, z4, z4, z8, 0);
#else
    (void)g; (void)lds_off; (void)tensor_rows; (void)row_len; (void)tile_rows;
#endif
}

// ---------------------------------------------------------------------------
__global__ void f32_to_bf16_kernel(const float* __restrict__ src,
                                   unsigned short* __restrict__ dst, int n) {
    int i = blockIdx.x * blockDim.x + threadIdx.x;
    if (i < n) dst[i] = f2bf(src[i]);
}

// ---------------------------------------------------------------------------
// LayerNorm over C=256 (one row per 256-thread block), bf16 output
// ---------------------------------------------------------------------------
__global__ __launch_bounds__(256) void ln_kernel(const float* __restrict__ x,
                                                 const float* __restrict__ g,
                                                 const float* __restrict__ b,
                                                 unsigned short* __restrict__ y) {
    __shared__ float red[256];
    int row = blockIdx.x, tid = threadIdx.x;
    float v = x[(size_t)row * CC + tid];
    red[tid] = v;
    __syncthreads();
    for (int s = 128; s > 0; s >>= 1) {
        if (tid < s) red[tid] += red[tid + s];
        __syncthreads();
    }
    float mean = red[0] * (1.f / CC);
    __syncthreads();
    float d = v - mean;
    red[tid] = d * d;
    __syncthreads();
    for (int s = 128; s > 0; s >>= 1) {
        if (tid < s) red[tid] += red[tid + s];
        __syncthreads();
    }
    float var = red[0] * (1.f / CC);
    y[(size_t)row * CC + tid] = f2bf(d * rsqrtf(var + 1e-5f) * g[tid] + b[tid]);
}

// ---------------------------------------------------------------------------
// Tiled bf16 WMMA GEMM: Out[M,N] = A[M,K] @ W[N,K]^T + bias, with epilogues.
// Block = 256 thr = 8 waves; block tile 128x64; wave tile 32x32 (4 accs).
// TDM double-buffers the LDS tiles: wave 0 issues next A/B DMAs, waits
// TENSORcnt<=2 (in-order per wave), barrier publishes the current tile.
// mode 0: store bf16 (qkv)         mode 1: +resid, store f32 (proj -> x1)
// mode 2: BN+GELU, store bf16(fc1) mode 3: BN+GELU+resid, store f32 (fc2)
// ---------------------------------------------------------------------------
__global__ __launch_bounds__(256) void gemm_bf16_kernel(
    const unsigned short* __restrict__ A, const unsigned short* __restrict__ Wt,
    const float* __restrict__ bias, void* __restrict__ out,
    const float* __restrict__ resid,
    const float* __restrict__ bng, const float* __restrict__ bnb,
    const float* __restrict__ bnm, const float* __restrict__ bnv,
    int M, int N, int K, int mode) {
    __shared__ unsigned short As[2][128 * 32];   // 8KB x2
    __shared__ unsigned short Bs[2][64 * 32];    // 4KB x2
    int tid  = threadIdx.x;
    int m0   = blockIdx.x * 128;
    int n0   = blockIdx.y * 64;
    int wave = tid >> 5, lane = tid & 31;
    int wm = wave & 3, wn = wave >> 2;           // 4 x 2 wave grid (32x32 each)

    v8f acc00, acc01, acc10, acc11;
    for (int i = 0; i < 8; ++i) {
        acc00[i] = 0.f; acc01[i] = 0.f; acc10[i] = 0.f; acc11[i] = 0.f;
    }

    int steps = K >> 5;
#if USE_TDM
    if (wave == 0) {
        tdm_load_tile_2d(&A[(size_t)m0 * K], (unsigned)(size_t)&As[0][0],
                         (unsigned)M, (unsigned)K, 128u);
        tdm_load_tile_2d(&Wt[(size_t)n0 * K], (unsigned)(size_t)&Bs[0][0],
                         (unsigned)N, (unsigned)K, 64u);
    }
#endif
    for (int i = 0; i < steps; ++i) {
        int buf = i & 1;
#if USE_TDM
        if (wave == 0) {
            if (i + 1 < steps) {
                int k1 = (i + 1) << 5;
                tdm_load_tile_2d(&A[(size_t)m0 * K + k1],
                                 (unsigned)(size_t)&As[buf ^ 1][0],
                                 (unsigned)M, (unsigned)K, 128u);
                tdm_load_tile_2d(&Wt[(size_t)n0 * K + k1],
                                 (unsigned)(size_t)&Bs[buf ^ 1][0],
                                 (unsigned)N, (unsigned)K, 64u);
                __builtin_amdgcn_s_wait_tensorcnt(2);  // current pair retired
            } else {
                __builtin_amdgcn_s_wait_tensorcnt(0);
            }
        }
        __syncthreads();
#else
        int k0 = i << 5;
        for (int v2 = 0; v2 < 2; ++v2) {         // A tile: 512 u32x4, 2/thread
            int j = tid * 2 + v2;
            int r = j >> 2, sg = (j & 3) * 8;
            *(u32x4*)&As[buf][r * 32 + sg] =
                *(const u32x4*)&A[(size_t)(m0 + r) * K + k0 + sg];
        }
        {                                        // B tile: 256 u32x4, 1/thread
            int r = tid >> 2, sg = (tid & 3) * 8;
            *(u32x4*)&Bs[buf][r * 32 + sg] =
                *(const u32x4*)&Wt[(size_t)(n0 + r) * K + k0 + sg];
        }
        __syncthreads();
#endif
        FragBF a0 = load_a_frag(As[buf], 32, wm * 32, 0, lane);
        FragBF a1 = load_a_frag(As[buf], 32, wm * 32 + 16, 0, lane);
        FragBF b0 = load_b_frag(Bs[buf], 32, wn * 32, 0, lane);
        FragBF b1 = load_b_frag(Bs[buf], 32, wn * 32 + 16, 0, lane);
        acc00 = WMMA_BF16(a0.v, b0.v, acc00);
        acc01 = WMMA_BF16(a0.v, b1.v, acc01);
        acc10 = WMMA_BF16(a1.v, b0.v, acc10);
        acc11 = WMMA_BF16(a1.v, b1.v, acc11);
        __syncthreads();
    }

    int nl    = lane & 15;
    int mhalf = (lane >> 4) * 8;                 // C/D layout: m = r + 8*(lane>=16)
    v8f* accs[2][2] = {{&acc00, &acc01}, {&acc10, &acc11}};
    for (int a = 0; a < 2; ++a) {
        for (int h2 = 0; h2 < 2; ++h2) {
            for (int r = 0; r < 8; ++r) {
                int row = m0 + wm * 32 + a * 16 + mhalf + r;
                int col = n0 + wn * 32 + h2 * 16 + nl;
                float v = (*accs[a][h2])[r] + bias[col];
                size_t o = (size_t)row * N + col;
                if (mode == 0) {
                    ((unsigned short*)out)[o] = f2bf(v);
                } else if (mode == 1) {
                    ((float*)out)[o] = v + resid[o];
                } else {
                    float aa = bng[col] * rsqrtf(bnv[col] + 1e-5f);
                    float vv = v * aa + (bnb[col] - bnm[col] * aa);
                    vv = gelu_f(vv);
                    if (mode == 2) ((unsigned short*)out)[o] = f2bf(vv);
                    else           ((float*)out)[o] = resid[o] + vv;
                }
            }
        }
    }
}

// ---------------------------------------------------------------------------
// Windowed attention: one wave per (window, head). Q*K^T via WMMA, fp32
// softmax with conf bias + 49-pad mask in LDS, P*V via WMMA.
// ---------------------------------------------------------------------------
__global__ __launch_bounds__(32) void attn_kernel(
    const unsigned short* __restrict__ qkv, const float* __restrict__ conf,
    unsigned short* __restrict__ obuf) {
    int head = blockIdx.x & 7;
    int win  = blockIdx.x >> 3;
    int b    = win >> 6;                   // 64 windows per image
    int w    = win & 63;
    int wh = w >> 3, ww = w & 7;

    __shared__ unsigned short Qs[64 * 32];
    __shared__ unsigned short Ks[64 * 32];
    __shared__ unsigned short Vt[32 * 64]; // transposed: [d][key]
    __shared__ unsigned short Ps[64 * 64]; // softmax probs, bf16
    __shared__ float          Ssf[64 * 64];
    __shared__ float          confs[64];

    int tid = threadIdx.x;
    for (int p = tid; p < 64; p += 32) {
        if (p < WS2) {
            int row = wh * WS + p / WS, col = ww * WS + p % WS;
            int n = row * WW2 + col;
            size_t base = ((size_t)(b * NN + n)) * (3 * CC) + head * HD;
            for (int d = 0; d < HD; d += 8) {
                *(u32x4*)&Qs[p * 32 + d] = *(const u32x4*)&qkv[base + d];
                *(u32x4*)&Ks[p * 32 + d] = *(const u32x4*)&qkv[base + CC + d];
            }
            for (int d = 0; d < HD; ++d) Vt[d * 64 + p] = qkv[base + 2 * CC + d];
            confs[p] = conf[b * NN + n];
        } else {
            u32x4 z = {0u, 0u, 0u, 0u};
            for (int d = 0; d < HD; d += 8) {
                *(u32x4*)&Qs[p * 32 + d] = z;
                *(u32x4*)&Ks[p * 32 + d] = z;
            }
            for (int d = 0; d < HD; ++d) Vt[d * 64 + p] = 0;
            confs[p] = 0.f;
        }
    }
    __syncthreads();

    const float scale = 0.17677669529663687f;  // HD^-0.5
    int lane  = tid;
    int nl    = lane & 15;
    int mhalf = (lane >> 4) * 8;

    for (int mt = 0; mt < 4; ++mt) {
        FragBF aq = load_a_frag(Qs, 32, mt * 16, 0, lane);
        for (int nt = 0; nt < 4; ++nt) {
            FragBF bk = load_b_frag(Ks, 32, nt * 16, 0, lane);
            v8f s;
            for (int i = 0; i < 8; ++i) s[i] = 0.f;
            s = WMMA_BF16(aq.v, bk.v, s);
            int   key = nt * 16 + nl;
            float bv  = (key < WS2) ? confs[key] : -1e30f;
            for (int r = 0; r < 8; ++r) {
                int m = mt * 16 + mhalf + r;
                Ssf[m * 64 + key] = (key < WS2) ? (s[r] * scale + bv) : -1e30f;
            }
        }
    }
    __syncthreads();

    for (int p = tid; p < 64; p += 32) {
        if (p < WS2) {
            float mx = -1e30f;
            for (int k = 0; k < WS2; ++k) mx = fmaxf(mx, Ssf[p * 64 + k]);
            float sum = 0.f;
            for (int k = 0; k < WS2; ++k) sum += __expf(Ssf[p * 64 + k] - mx);
            float inv = 1.f / sum;
            for (int k = 0; k < WS2; ++k)
                Ps[p * 64 + k] = f2bf(__expf(Ssf[p * 64 + k] - mx) * inv);
            for (int k = WS2; k < 64; ++k) Ps[p * 64 + k] = 0;
        } else {
            for (int k = 0; k < 64; ++k) Ps[p * 64 + k] = 0;
        }
    }
    __syncthreads();

    for (int mt = 0; mt < 4; ++mt) {
        FragBF ap0 = load_a_frag(Ps, 64, mt * 16, 0, lane);
        FragBF ap1 = load_a_frag(Ps, 64, mt * 16, 32, lane);
        for (int ntv = 0; ntv < 2; ++ntv) {
            FragBF bv0 = load_b_frag(Vt, 64, ntv * 16, 0, lane);
            FragBF bv1 = load_b_frag(Vt, 64, ntv * 16, 32, lane);
            v8f oacc;
            for (int i = 0; i < 8; ++i) oacc[i] = 0.f;
            oacc = WMMA_BF16(ap0.v, bv0.v, oacc);
            oacc = WMMA_BF16(ap1.v, bv1.v, oacc);
            for (int r = 0; r < 8; ++r) {
                int p = mt * 16 + mhalf + r;
                if (p < WS2) {
                    int row = wh * WS + p / WS, col = ww * WS + p % WS;
                    int n = row * WW2 + col;
                    obuf[((size_t)(b * NN + n)) * CC + head * HD + ntv * 16 + nl] =
                        f2bf(oacc[r]);
                }
            }
        }
    }
}

// ---------------------------------------------------------------------------
// token2map scatter (atomic mean-pool into H x W map)
// ---------------------------------------------------------------------------
__global__ __launch_bounds__(256) void scatter_kernel(
    const unsigned short* __restrict__ h, const float* __restrict__ loc,
    const int* __restrict__ idx_agg, float* __restrict__ fmap,
    float* __restrict__ cnt) {
    int pt = blockIdx.x;
    int b  = pt / NN;
    float lx = loc[(size_t)pt * 2 + 0], ly = loc[(size_t)pt * 2 + 1];
    int px = min(max((int)floorf(lx * (float)WW2), 0), WW2 - 1);
    int py = min(max((int)floorf(ly * (float)HH), 0), HH - 1);
    int hw  = py * WW2 + px;
    int idx = idx_agg[pt];
    size_t src = ((size_t)b * NN + idx) * HID;
    size_t dst = ((size_t)b * NN + hw) * HID;
    for (int c = threadIdx.x; c < HID; c += 256)
        atomicAdd(&fmap[dst + c], bf2f(h[src + c]));
    if (threadIdx.x == 0) atomicAdd(&cnt[b * NN + hw], 1.f);
}

__global__ __launch_bounds__(256) void map_norm_kernel(float* __restrict__ fmap,
                                                       const float* __restrict__ cnt) {
    int pix = blockIdx.x;
    float inv = 1.f / (cnt[pix] + 1e-6f);
    size_t base = (size_t)pix * HID;
    for (int c = threadIdx.x; c < HID; c += 256) fmap[base + c] *= inv;
}

// ---------------------------------------------------------------------------
// Fused 3x3 depthwise conv + map2token gather (atomic weighted mean)
// ---------------------------------------------------------------------------
__global__ __launch_bounds__(256) void conv_gather_kernel(
    const float* __restrict__ fmap, const float* __restrict__ dww,
    const float* __restrict__ dwb, const float* __restrict__ loc,
    const int* __restrict__ idx_agg, const float* __restrict__ aggw,
    float* __restrict__ num, float* __restrict__ den) {
    int pt = blockIdx.x;
    int b  = pt / NN;
    float lx = loc[(size_t)pt * 2 + 0], ly = loc[(size_t)pt * 2 + 1];
    int px = min(max((int)floorf(lx * (float)WW2), 0), WW2 - 1);
    int py = min(max((int)floorf(ly * (float)HH), 0), HH - 1);
    float wgt = aggw[pt];
    int   idx = idx_agg[pt];
    size_t tok = ((size_t)b * NN + idx) * HID;
    for (int c = threadIdx.x; c < HID; c += 256) {
        float s = dwb[c];
        for (int ky = 0; ky < 3; ++ky) {
            int yy = py + ky - 1;
            if (yy < 0 || yy >= HH) continue;
            for (int kx = 0; kx < 3; ++kx) {
                int xx = px + kx - 1;
                if (xx < 0 || xx >= WW2) continue;
                s += fmap[((size_t)b * NN + yy * WW2 + xx) * HID + c] *
                     dww[(ky * 3 + kx) * HID + c];
            }
        }
        atomicAdd(&num[tok + c], s * wgt);
    }
    if (threadIdx.x == 0) atomicAdd(&den[b * NN + idx], wgt);
}

// ---------------------------------------------------------------------------
// merge: g = GELU(BN2(h * dws + num/(den+eps))), bf16 out
// ---------------------------------------------------------------------------
__global__ __launch_bounds__(256) void merge_kernel(
    const unsigned short* __restrict__ h, const float* __restrict__ num,
    const float* __restrict__ den, const float* __restrict__ dws,
    const float* __restrict__ g2, const float* __restrict__ b2,
    const float* __restrict__ m2, const float* __restrict__ v2,
    unsigned short* __restrict__ gout) {
    int t = blockIdx.x;
    float inv = 1.f / (den[t] + 1e-6f);
    size_t base = (size_t)t * HID;
    for (int c = threadIdx.x; c < HID; c += 256) {
        float hv = bf2f(h[base + c]) * dws[c] + num[base + c] * inv;
        float a  = g2[c] * rsqrtf(v2[c] + 1e-5f);
        hv = hv * a + (b2[c] - m2[c] * a);
        gout[base + c] = f2bf(gelu_f(hv));
    }
}

// ---------------------------------------------------------------------------
extern "C" void kernel_launch(void* const* d_in, const int* in_sizes, int n_in,
                              void* d_out, int out_size, void* d_ws,
                              size_t ws_size, hipStream_t stream) {
    (void)in_sizes; (void)n_in; (void)out_size; (void)ws_size;

    const float* x      = (const float*)d_in[0];
    const float* conf   = (const float*)d_in[1];
    const float* loc    = (const float*)d_in[2];
    const float* aggw   = (const float*)d_in[3];
    const int*   idxagg = (const int*)d_in[4];
    const float* ln1_g  = (const float*)d_in[7];
    const float* ln1_b  = (const float*)d_in[8];
    const float* q_w    = (const float*)d_in[9];
    const float* q_b    = (const float*)d_in[10];
    const float* kv_w   = (const float*)d_in[11];
    const float* kv_b   = (const float*)d_in[12];
    const float* proj_w = (const float*)d_in[13];
    const float* proj_b = (const float*)d_in[14];
    const float* ln2_g  = (const float*)d_in[15];
    const float* ln2_b  = (const float*)d_in[16];
    const float* fc1_w  = (const float*)d_in[17];
    const float* fc1_b  = (const float*)d_in[18];
    const float* bn1_g  = (const float*)d_in[19];
    const float* bn1_b  = (const float*)d_in[20];
    const float* bn1_m  = (const float*)d_in[21];
    const float* bn1_v  = (const float*)d_in[22];
    const float* dw_w   = (const float*)d_in[23];
    const float* dw_b   = (const float*)d_in[24];
    const float* dws_w  = (const float*)d_in[25];
    const float* bn2_g  = (const float*)d_in[26];
    const float* bn2_b  = (const float*)d_in[27];
    const float* bn2_m  = (const float*)d_in[28];
    const float* bn2_v  = (const float*)d_in[29];
    const float* fc2_w  = (const float*)d_in[30];
    const float* fc2_b  = (const float*)d_in[31];
    const float* bn3_g  = (const float*)d_in[32];
    const float* bn3_b  = (const float*)d_in[33];
    const float* bn3_m  = (const float*)d_in[34];
    const float* bn3_v  = (const float*)d_in[35];
    float* outp = (float*)d_out;

    char*  ws  = (char*)d_ws;
    size_t off = 0;
    auto alloc = [&](size_t bytes) -> void* {
        void* p = (void*)(ws + off);
        off = (off + bytes + 255) & ~(size_t)255;
        return p;
    };
    unsigned short* wqkv  = (unsigned short*)alloc((size_t)3 * CC * CC * 2);
    unsigned short* wproj = (unsigned short*)alloc((size_t)CC * CC * 2);
    unsigned short* wfc1  = (unsigned short*)alloc((size_t)HID * CC * 2);
    unsigned short* wfc2  = (unsigned short*)alloc((size_t)CC * HID * 2);
    float*          bqkv  = (float*)alloc((size_t)3 * CC * 4);
    unsigned short* yb    = (unsigned short*)alloc((size_t)NTOK * CC * 2);
    unsigned short* qkv   = (unsigned short*)alloc((size_t)NTOK * 3 * CC * 2);
    unsigned short* obuf  = (unsigned short*)alloc((size_t)NTOK * CC * 2);
    float*          x1    = (float*)alloc((size_t)NTOK * CC * 4);
    unsigned short* zb    = (unsigned short*)alloc((size_t)NTOK * CC * 2);
    unsigned short* hb    = (unsigned short*)alloc((size_t)NTOK * HID * 2);
    float*          fmap  = (float*)alloc((size_t)NTOK * HID * 4);
    float*          cnt   = (float*)alloc((size_t)NTOK * 4);
    float*          num   = (float*)alloc((size_t)NTOK * HID * 4);
    float*          den   = (float*)alloc((size_t)NTOK * 4);
    unsigned short* gout  = (unsigned short*)fmap;   // reuse fmap region

    // weights -> bf16 (qkv fused: [q_w; kv_w] = [768, 256])
    f32_to_bf16_kernel<<<(CC * CC + 255) / 256, 256, 0, stream>>>(q_w, wqkv, CC * CC);
    f32_to_bf16_kernel<<<(2 * CC * CC + 255) / 256, 256, 0, stream>>>(
        kv_w, wqkv + CC * CC, 2 * CC * CC);
    f32_to_bf16_kernel<<<(CC * CC + 255) / 256, 256, 0, stream>>>(proj_w, wproj, CC * CC);
    f32_to_bf16_kernel<<<(HID * CC + 255) / 256, 256, 0, stream>>>(fc1_w, wfc1, HID * CC);
    f32_to_bf16_kernel<<<(CC * HID + 255) / 256, 256, 0, stream>>>(fc2_w, wfc2, CC * HID);
    (void)hipMemcpyAsync(bqkv, q_b, CC * 4, hipMemcpyDeviceToDevice, stream);
    (void)hipMemcpyAsync(bqkv + CC, kv_b, 2 * CC * 4, hipMemcpyDeviceToDevice, stream);

    // LN1 -> yb
    ln_kernel<<<NTOK, 256, 0, stream>>>(x, ln1_g, ln1_b, yb);
    // fused QKV GEMM -> qkv (bf16)
    gemm_bf16_kernel<<<dim3(NTOK / 128, (3 * CC) / 64), 256, 0, stream>>>(
        yb, wqkv, bqkv, qkv, nullptr, nullptr, nullptr, nullptr, nullptr,
        NTOK, 3 * CC, CC, 0);
    // windowed attention
    attn_kernel<<<BB * 64 * HEADS, 32, 0, stream>>>(qkv, conf, obuf);
    // proj + residual -> x1 (f32)
    gemm_bf16_kernel<<<dim3(NTOK / 128, CC / 64), 256, 0, stream>>>(
        obuf, wproj, proj_b, x1, x, nullptr, nullptr, nullptr, nullptr,
        NTOK, CC, CC, 1);
    // LN2 -> zb
    ln_kernel<<<NTOK, 256, 0, stream>>>(x1, ln2_g, ln2_b, zb);
    // fc1 + BN1 + GELU -> hb (bf16)
    gemm_bf16_kernel<<<dim3(NTOK / 128, HID / 64), 256, 0, stream>>>(
        zb, wfc1, fc1_b, hb, nullptr, bn1_g, bn1_b, bn1_m, bn1_v,
        NTOK, HID, CC, 2);
    // token2map
    (void)hipMemsetAsync(fmap, 0, (size_t)NTOK * HID * 4, stream);
    (void)hipMemsetAsync(cnt, 0, (size_t)NTOK * 4, stream);
    scatter_kernel<<<NTOK, 256, 0, stream>>>(hb, loc, idxagg, fmap, cnt);
    map_norm_kernel<<<NTOK, 256, 0, stream>>>(fmap, cnt);
    // fused depthwise conv + map2token
    (void)hipMemsetAsync(num, 0, (size_t)NTOK * HID * 4, stream);
    (void)hipMemsetAsync(den, 0, (size_t)NTOK * 4, stream);
    conv_gather_kernel<<<NTOK, 256, 0, stream>>>(fmap, dw_w, dw_b, loc, idxagg,
                                                 aggw, num, den);
    // merge + BN2 + GELU -> gout (bf16, reuses fmap region)
    merge_kernel<<<NTOK, 256, 0, stream>>>(hb, num, den, dws_w, bn2_g, bn2_b,
                                           bn2_m, bn2_v, gout);
    // fc2 + BN3 + GELU + residual -> d_out (f32)
    gemm_bf16_kernel<<<dim3(NTOK / 128, CC / 64), 256, 0, stream>>>(
        gout, wfc2, fc2_b, outp, x1, bn3_g, bn3_b, bn3_m, bn3_v,
        NTOK, CC, HID, 3);
}
